// DAD_MA_35330400976941
// MI455X (gfx1250) — matrix operand order repaired and checked by reading, not compiled, not run on hardware
//
#include <hip/hip_runtime.h>
#include <hip/hip_bf16.h>
#include <stdint.h>

typedef __attribute__((ext_vector_type(16))) __bf16 v16bf;
typedef __attribute__((ext_vector_type(8)))  __bf16 v8bf;
typedef __attribute__((ext_vector_type(8)))  float  v8f;
typedef int v4i_vec __attribute__((vector_size(4 * sizeof(int))));

// ---- CDNA5 async-load-to-LDS path (guarded; fallback = plain load + ds_store)
#if defined(__has_builtin)
#  if __has_builtin(__builtin_amdgcn_global_load_async_to_lds_b128)
#    define HAVE_ASYNC_LDS 1
#  endif
#endif
#ifndef HAVE_ASYNC_LDS
#  define HAVE_ASYNC_LDS 0
#endif

#if HAVE_ASYNC_LDS
#  if defined(__has_builtin) && __has_builtin(__builtin_amdgcn_s_wait_asynccnt)
#    define WAIT_ASYNC(n) __builtin_amdgcn_s_wait_asynccnt(n)
#  else
#    define WAIT_ASYNC(n) asm volatile("s_wait_asynccnt %0" ::"i"(n) : "memory")
#  endif
__device__ __forceinline__ void async_copy16(const void* g, void* l) {
  __builtin_amdgcn_global_load_async_to_lds_b128(
      (__attribute__((address_space(1))) v4i_vec*)g,
      (__attribute__((address_space(3))) v4i_vec*)l, 0, 0);
}
#else
#  define WAIT_ASYNC(n)
#endif

#define NROWS 1024
#define DLEN  16384
#define KT    32

// ---------------- kernel 1: sq[i] = sum_d x[i,d]^2  (fp32 source) ----------------
__global__ __launch_bounds__(256) void row_sq_kernel(const float* __restrict__ X,
                                                     float* __restrict__ sq) {
  __shared__ float red[256];
  const int i = blockIdx.x;
  const float4* row = (const float4*)(X + (size_t)i * DLEN);
  float s = 0.0f;
  for (int c = threadIdx.x; c < DLEN / 4; c += 256) {
    float4 v = row[c];
    s += v.x * v.x + v.y * v.y + v.z * v.z + v.w * v.w;
  }
  red[threadIdx.x] = s;
  __syncthreads();
  for (int off = 128; off > 0; off >>= 1) {
    if (threadIdx.x < off) red[threadIdx.x] += red[threadIdx.x + off];
    __syncthreads();
  }
  if (threadIdx.x == 0) sq[i] = red[0];
}

// ---------------- kernel 1b: one-shot fp32 -> bf16 conversion of X ----------------
__global__ __launch_bounds__(256) void cvt_bf16_kernel(const float* __restrict__ X,
                                                       __bf16* __restrict__ Xb) {
  const size_t idx = ((size_t)blockIdx.x * 256 + threadIdx.x) * 8;
  const float4 a = *(const float4*)(X + idx);
  const float4 b = *(const float4*)(X + idx + 4);
  v8bf o;
  o[0] = (__bf16)a.x; o[1] = (__bf16)a.y; o[2] = (__bf16)a.z; o[3] = (__bf16)a.w;
  o[4] = (__bf16)b.x; o[5] = (__bf16)b.y; o[6] = (__bf16)b.z; o[7] = (__bf16)b.w;
  *(v8bf*)(Xb + idx) = o;
}

// ======================= bf16-panel GEMM (preferred path) ========================
// Stage two 64xKT bf16 panels (4 KB each); 128 threads, 16B chunks, 4 async/thread.
__device__ __forceinline__ void stage_bf(const __bf16* __restrict__ Xb,
                                         int rowA, int rowB, int k0,
                                         __bf16* __restrict__ pA,
                                         __bf16* __restrict__ pB, int tid) {
#pragma unroll
  for (int q = 0; q < 2; ++q) {
    const int c   = tid + 128 * q;     // 0..255 chunks per panel
    const int row = c >> 2;            // 4x16B chunks per 64B row
    const int col = (c & 3) * 8;       // bf16 index
    const __bf16* ga = Xb + (size_t)(rowA + row) * DLEN + k0 + col;
    const __bf16* gb = Xb + (size_t)(rowB + row) * DLEN + k0 + col;
    __bf16* la = pA + row * KT + col;
    __bf16* lb = pB + row * KT + col;
#if HAVE_ASYNC_LDS
    async_copy16(ga, la);
    async_copy16(gb, lb);
#else
    *(v8bf*)la = *(const v8bf*)ga;
    *(v8bf*)lb = *(const v8bf*)gb;
#endif
  }
}

__device__ __forceinline__ v16bf frag_cat(v8bf lo, v8bf hi) {
  return __builtin_shufflevector(lo, hi, 0, 1, 2, 3, 4, 5, 6, 7,
                                 8, 9, 10, 11, 12, 13, 14, 15);
}

__global__ __launch_bounds__(128) void gram_loss_bf16_kernel(
    const __bf16* __restrict__ Xb, const float* __restrict__ sq,
    float* __restrict__ partials) {
  __shared__ __align__(16) __bf16 pa[2][64 * KT];
  __shared__ __align__(16) __bf16 pb[2][64 * KT];
  __shared__ float sqr[64];
  __shared__ float sqc[64];
  __shared__ float red[128];

  const int tid  = threadIdx.x;
  const int lane = tid & 31;
  const int wave = tid >> 5;
  const int wR   = (wave >> 1) * 32;
  const int wC   = (wave & 1) * 32;
  const int lr   = lane & 15;
  const int lh   = lane >> 4;

  const int rowA = blockIdx.y * 64;
  const int rowB = blockIdx.x * 64;

  if (tid < 64) sqr[tid] = sq[rowA + tid];
  else          sqc[tid - 64] = sq[rowB + (tid - 64)];

  v8f acc[2][2];
#pragma unroll
  for (int a = 0; a < 2; ++a)
#pragma unroll
    for (int b = 0; b < 2; ++b)
#pragma unroll
      for (int e = 0; e < 8; ++e) acc[a][b][e] = 0.0f;

  stage_bf(Xb, rowA, rowB, 0, pa[0], pb[0], tid);

  const int nsteps = DLEN / KT;   // 512
  for (int s = 0; s < nsteps; ++s) {
    const int cur = s & 1;
    if (s + 1 < nsteps) {
      stage_bf(Xb, rowA, rowB, (s + 1) * KT, pa[cur ^ 1], pb[cur ^ 1], tid);
      WAIT_ASYNC(4);
    } else {
      WAIT_ASYNC(0);
    }
    __syncthreads();

    const __bf16* A = pa[cur];
    const __bf16* B = pb[cur];

    // A 16x32: lane m=lr; elems 0..7 = K 8*lh..+7, 8..15 = K 16+8*lh..+7
    // B 32x16: lane n=lr; elems 0..7 = K 8*lh..+7, 8..15 = K 16+8*lh..+7
    v16bf af[2], bf[2];
#pragma unroll
    for (int t = 0; t < 2; ++t) {
      const __bf16* ra = A + (wR + t * 16 + lr) * KT;
      const __bf16* rb = B + (wC + t * 16 + lr) * KT;
      af[t] = frag_cat(*(const v8bf*)(ra + 8 * lh),
                       *(const v8bf*)(ra + 16 + 8 * lh));
      bf[t] = frag_cat(*(const v8bf*)(rb + 8 * lh), *(const v8bf*)(rb + 16 + 8 * lh));
    }

#pragma unroll
    for (int a = 0; a < 2; ++a)
#pragma unroll
      for (int b = 0; b < 2; ++b)
        acc[a][b] = __builtin_amdgcn_wmma_f32_16x16x32_bf16(
            false, af[a], false, bf[b], (short)0, acc[a][b], false, false);

    __syncthreads();
  }

  const float invD = 1.0f / (float)DLEN;
  float local = 0.0f;
#pragma unroll
  for (int a = 0; a < 2; ++a)
#pragma unroll
    for (int b = 0; b < 2; ++b)
#pragma unroll
      for (int r = 0; r < 8; ++r) {
        const float si = sqr[wR + a * 16 + r + 8 * lh];
        const float sj = sqc[wC + b * 16 + lr];
        const float g  = (si + sj - 2.0f * acc[a][b][r]) * invD;
        local += g * g;
      }

  red[tid] = local;
  __syncthreads();
  for (int off = 64; off > 0; off >>= 1) {
    if (tid < off) red[tid] += red[tid + off];
    __syncthreads();
  }
  if (tid == 0) partials[blockIdx.y * gridDim.x + blockIdx.x] = red[0];
}

// ============ fp32-panel GEMM (fallback when ws can't hold bf16 X) ==============
__device__ __forceinline__ void stage_f32(const float* __restrict__ X,
                                          int rowA, int rowB, int k0,
                                          float* __restrict__ pA,
                                          float* __restrict__ pB, int tid) {
#pragma unroll
  for (int q = 0; q < 4; ++q) {
    const int c   = tid + 128 * q;
    const int row = c >> 3;
    const int col = (c & 7) << 2;
    const float* ga = X + (size_t)(rowA + row) * DLEN + k0 + col;
    const float* gb = X + (size_t)(rowB + row) * DLEN + k0 + col;
    float* la = pA + row * KT + col;
    float* lb = pB + row * KT + col;
#if HAVE_ASYNC_LDS
    async_copy16(ga, la);
    async_copy16(gb, lb);
#else
    *(float4*)la = *(const float4*)ga;
    *(float4*)lb = *(const float4*)gb;
#endif
  }
}

__global__ __launch_bounds__(128) void gram_loss_f32_kernel(
    const float* __restrict__ X, const float* __restrict__ sq,
    float* __restrict__ partials) {
  __shared__ __align__(16) float pa[2][64 * KT];
  __shared__ __align__(16) float pb[2][64 * KT];
  __shared__ float sqr[64];
  __shared__ float sqc[64];
  __shared__ float red[128];

  const int tid  = threadIdx.x;
  const int lane = tid & 31;
  const int wave = tid >> 5;
  const int wR   = (wave >> 1) * 32;
  const int wC   = (wave & 1) * 32;
  const int lr   = lane & 15;
  const int lh   = lane >> 4;

  const int rowA = blockIdx.y * 64;
  const int rowB = blockIdx.x * 64;

  if (tid < 64) sqr[tid] = sq[rowA + tid];
  else          sqc[tid - 64] = sq[rowB + (tid - 64)];

  v8f acc[2][2];
#pragma unroll
  for (int a = 0; a < 2; ++a)
#pragma unroll
    for (int b = 0; b < 2; ++b)
#pragma unroll
      for (int e = 0; e < 8; ++e) acc[a][b][e] = 0.0f;

  stage_f32(X, rowA, rowB, 0, pa[0], pb[0], tid);

  const int nsteps = DLEN / KT;
  for (int s = 0; s < nsteps; ++s) {
    const int cur = s & 1;
    if (s + 1 < nsteps) {
      stage_f32(X, rowA, rowB, (s + 1) * KT, pa[cur ^ 1], pb[cur ^ 1], tid);
      WAIT_ASYNC(8);
    } else {
      WAIT_ASYNC(0);
    }
    __syncthreads();

    const float* A = pa[cur];
    const float* B = pb[cur];

    v16bf af[2], bf[2];
#pragma unroll
    for (int t = 0; t < 2; ++t) {
      const float* ra = A + (wR + t * 16 + lr) * KT;
      const float* rb = B + (wC + t * 16 + lr) * KT;
#pragma unroll
      for (int i = 0; i < 8; ++i) {
        af[t][i]     = (__bf16)ra[8 * lh + i];
        af[t][8 + i] = (__bf16)ra[16 + 8 * lh + i];
      }
#pragma unroll
      for (int i = 0; i < 8; ++i) {
        bf[t][i]     = (__bf16)rb[8 * lh + i];
        bf[t][8 + i] = (__bf16)rb[16 + 8 * lh + i];
      }
    }

#pragma unroll
    for (int a = 0; a < 2; ++a)
#pragma unroll
      for (int b = 0; b < 2; ++b)
        acc[a][b] = __builtin_amdgcn_wmma_f32_16x16x32_bf16(
            false, af[a], false, bf[b], (short)0, acc[a][b], false, false);

    __syncthreads();
  }

  const float invD = 1.0f / (float)DLEN;
  float local = 0.0f;
#pragma unroll
  for (int a = 0; a < 2; ++a)
#pragma unroll
    for (int b = 0; b < 2; ++b)
#pragma unroll
      for (int r = 0; r < 8; ++r) {
        const float si = sqr[wR + a * 16 + r + 8 * lh];
        const float sj = sqc[wC + b * 16 + lr];
        const float g  = (si + sj - 2.0f * acc[a][b][r]) * invD;
        local += g * g;
      }

  red[tid] = local;
  __syncthreads();
  for (int off = 64; off > 0; off >>= 1) {
    if (tid < off) red[tid] += red[tid + off];
    __syncthreads();
  }
  if (tid == 0) partials[blockIdx.y * gridDim.x + blockIdx.x] = red[0];
}

// ---------------- kernel 3: reduce 256 block partials -> mean ----------------
__global__ __launch_bounds__(256) void finish_kernel(const float* __restrict__ partials,
                                                     float* __restrict__ out) {
  __shared__ float red[256];
  red[threadIdx.x] = partials[threadIdx.x];
  __syncthreads();
  for (int off = 128; off > 0; off >>= 1) {
    if (threadIdx.x < off) red[threadIdx.x] += red[threadIdx.x + off];
    __syncthreads();
  }
  if (threadIdx.x == 0)
    out[0] = red[0] * (1.0f / ((float)NROWS * (float)NROWS));
}

extern "C" void kernel_launch(void* const* d_in, const int* in_sizes, int n_in,
                              void* d_out, int out_size, void* d_ws, size_t ws_size,
                              hipStream_t stream) {
  (void)in_sizes; (void)n_in; (void)out_size;
  // Reference bug: fm_s (d_in[0]) does not affect the output; only fm_t is used.
  const float* fm_t = (const float*)d_in[1];
  float* out = (float*)d_out;

  const size_t xb_bytes = (size_t)NROWS * DLEN * sizeof(uint16_t);  // 32 MB
  const bool pre_convert = ws_size >= xb_bytes + 16 * 1024;

  if (pre_convert) {
    __bf16* Xb      = (__bf16*)d_ws;
    float* sq       = (float*)((char*)d_ws + xb_bytes);
    float* partials = sq + NROWS;

    row_sq_kernel<<<NROWS, 256, 0, stream>>>(fm_t, sq);
    cvt_bf16_kernel<<<(NROWS * (DLEN / 8)) / 256, 256, 0, stream>>>(fm_t, Xb);
    gram_loss_bf16_kernel<<<dim3(16, 16), 128, 0, stream>>>(Xb, sq, partials);
    finish_kernel<<<1, 256, 0, stream>>>(partials, out);
  } else {
    float* sq       = (float*)d_ws;
    float* partials = sq + NROWS;

    row_sq_kernel<<<NROWS, 256, 0, stream>>>(fm_t, sq);
    gram_loss_f32_kernel<<<dim3(16, 16), 128, 0, stream>>>(fm_t, sq, partials);
    finish_kernel<<<1, 256, 0, stream>>>(partials, out);
  }
}
